// MultiheadAttention_88802743812872
// MI455X (gfx1250) — matrix operand order repaired and checked by reading, not compiled, not run on hardware
//
#include <hip/hip_runtime.h>
#include <hip/hip_bf16.h>

typedef __attribute__((ext_vector_type(16))) _Float16 v16h;
typedef __attribute__((ext_vector_type(8)))  _Float16 v8h;
typedef __attribute__((ext_vector_type(8)))  float    v8f;

#define BB 4
#define SS 2048
#define EE 1024
#define HH 16
#define DK 64
#define MM (BB * SS)   // 8192 rows
#define N3 (3 * EE)    // 3072 qkv cols

// ---------------- WMMA fragment helpers (wave32, gfx1250) ----------------

__device__ __forceinline__ v8f wmma16(v16h a, v16h b, v8f c) {
  // D = A(16x32 f16) * B(32x16 f16) + C(16x16 f32)
  return __builtin_amdgcn_wmma_f32_16x16x32_f16(
      /*neg_a=*/false, a, /*neg_b=*/false, b,
      /*c_mod=*/(short)0, c, /*reuse_a=*/false, /*reuse_b=*/false);
}

__device__ __forceinline__ void pf(const void* p) {
  // lowers to global_prefetch_b8 on gfx1250
  __builtin_prefetch(p, 0, 1);
}

// A-matrix 16x32 f16, row-major source with leading dim lda (elements).
// ISA layout: lanes 0-15 (M=lane): halves 0-7 = K 0..7, halves 8-15 = K 16..23
//             lanes 16-31 (M=lane-16): K 8..15 and K 24..31.
__device__ __forceinline__ v16h load_a16(const _Float16* __restrict__ A,
                                         int row0, int k0, int lda) {
  int lane = threadIdx.x & 31;
  int m = lane & 15;
  int kb = (lane >> 4) ? 8 : 0;
  const _Float16* p = A + (size_t)(row0 + m) * lda + k0 + kb;
  v8h lo = *(const v8h*)p;          // K = kb .. kb+7
  v8h hi = *(const v8h*)(p + 16);   // K = kb+16 .. kb+23
  v16h r;
#pragma unroll
  for (int i = 0; i < 8; ++i) { r[i] = lo[i]; r[i + 8] = hi[i]; }
  return r;
}

// Same A fragment but source is fp32 (converted on load). lda in elements.
__device__ __forceinline__ v16h load_a32(const float* __restrict__ A,
                                         int row0, int k0, int lda) {
  int lane = threadIdx.x & 31;
  int m = lane & 15;
  int kb = (lane >> 4) ? 8 : 0;
  const float* p = A + (size_t)(row0 + m) * lda + k0 + kb;
  const float4* p4 = (const float4*)p;
  float4 x0 = p4[0], x1 = p4[1];   // K = kb .. kb+7
  float4 x2 = p4[4], x3 = p4[5];   // K = kb+16 .. kb+23
  v16h r;
  r[0]  = (_Float16)x0.x; r[1]  = (_Float16)x0.y; r[2]  = (_Float16)x0.z; r[3]  = (_Float16)x0.w;
  r[4]  = (_Float16)x1.x; r[5]  = (_Float16)x1.y; r[6]  = (_Float16)x1.z; r[7]  = (_Float16)x1.w;
  r[8]  = (_Float16)x2.x; r[9]  = (_Float16)x2.y; r[10] = (_Float16)x2.z; r[11] = (_Float16)x2.w;
  r[12] = (_Float16)x3.x; r[13] = (_Float16)x3.y; r[14] = (_Float16)x3.z; r[15] = (_Float16)x3.w;
  return r;
}

// B-matrix 32x16 f16 from an N-major (transposed) source: BT[n][k], ldb = K stride.
// ISA layout: lane n = lane%16; lanes 0-15 hold K 0..7,16..23, lanes 16-31 hold K 8..15,24..31.
__device__ __forceinline__ v16h load_b16(const _Float16* __restrict__ BT,
                                         int n0, int k0, int ldb) {
  int lane = threadIdx.x & 31;
  int n = lane & 15;
  int kb = (lane >> 4) ? 8 : 0;
  const _Float16* p = BT + (size_t)(n0 + n) * ldb + k0 + kb;
  v8h lo = *(const v8h*)p;
  v8h hi = *(const v8h*)(p + 16);
  v16h r;
#pragma unroll
  for (int i = 0; i < 8; ++i) { r[i] = lo[i]; r[i + 8] = hi[i]; }
  return r;
}

// ---------------- Precision conversion / layout kernels ----------------

__global__ void cvt_f32_f16(const float* __restrict__ s, _Float16* __restrict__ d, int n) {
  int i = blockIdx.x * blockDim.x + threadIdx.x;
  int stride = gridDim.x * blockDim.x;
  for (; i < n; i += stride) d[i] = (_Float16)s[i];
}

// W: [K,N] row-major fp32  ->  WT: [N,K] row-major f16
__global__ void transpose_f32_f16(const float* __restrict__ W, _Float16* __restrict__ WT,
                                  int K, int N) {
  int i = blockIdx.x * blockDim.x + threadIdx.x;
  int total = K * N;
  int stride = gridDim.x * blockDim.x;
  for (; i < total; i += stride) {
    int nn = i / K;
    int kk = i - nn * K;
    WT[i] = (_Float16)W[(size_t)kk * N + nn];
  }
}

// ---------------- QKV projection: X[8192,1024] @ Wqkv -> Q,K,Vt -------------
// Each wave: two 16-row M tiles x four 16-col N tiles (32x64 output),
// K loop 1024/32: 2 A-frags + 4 B-frags (all individually named so the
// scheduler can issue one load clause + partial s_wait_loadcnt) -> 8 WMMAs.
// Column f of qkv maps to head h=f/192, type t=(f%192)/64, dim d=f%64 (per the
// reference's reshape(B,S,H,3*dk) + split).

__global__ void __launch_bounds__(256) qkv_gemm(
    const _Float16* __restrict__ Xh, const _Float16* __restrict__ WT,
    const float* __restrict__ bias,
    _Float16* __restrict__ Qh,   // [B,H,S,DK]
    _Float16* __restrict__ Kh,   // [B,H,S,DK]
    _Float16* __restrict__ Vt) { // [B,H,DK,S]
  int wave = threadIdx.x >> 5;
  int gid = blockIdx.x * 8 + wave;
  int mt = gid / (N3 / 64);            // 0..255, 32-row tiles
  int ng = gid - mt * (N3 / 64);       // 0..47, 64-col groups
  int row0 = mt * 32;
  v8f c00 = {}, c01 = {}, c02 = {}, c03 = {};
  v8f c10 = {}, c11 = {}, c12 = {}, c13 = {};
  for (int k0 = 0; k0 < EE; k0 += 32) {
    v16h a0 = load_a16(Xh, row0,      k0, EE);
    v16h a1 = load_a16(Xh, row0 + 16, k0, EE);
    v16h b0 = load_b16(WT, ng * 64 +  0, k0, EE);
    v16h b1 = load_b16(WT, ng * 64 + 16, k0, EE);
    v16h b2 = load_b16(WT, ng * 64 + 32, k0, EE);
    v16h b3 = load_b16(WT, ng * 64 + 48, k0, EE);
    if (k0 + 256 < EE) {
      int lane = threadIdx.x & 31;
      pf(Xh + (size_t)(row0 + (lane & 15)) * EE + k0 + 256);
      pf(WT + (size_t)(ng * 64 + lane) * EE + k0 + 256);
    }
    c00 = wmma16(a0, b0, c00);
    c10 = wmma16(a1, b0, c10);
    c01 = wmma16(a0, b1, c01);
    c11 = wmma16(a1, b1, c11);
    c02 = wmma16(a0, b2, c02);
    c12 = wmma16(a1, b2, c12);
    c03 = wmma16(a0, b3, c03);
    c13 = wmma16(a1, b3, c13);
  }
  int lane = threadIdx.x & 31;
  int nl = lane & 15, hi = lane >> 4;
  v8f acc[2][4] = {{c00, c01, c02, c03}, {c10, c11, c12, c13}};
#pragma unroll
  for (int j = 0; j < 4; ++j) {
    int f = ng * 64 + j * 16 + nl;
    float bv = bias[f];
    int h = f / 192;
    int rem = f - h * 192;
    int t = rem / 64;
    int d = rem - t * 64;
#pragma unroll
    for (int i = 0; i < 2; ++i) {
#pragma unroll
      for (int r = 0; r < 8; ++r) {
        int m = row0 + i * 16 + r + 8 * hi;
        int b = m / SS;
        int s = m - b * SS;
        size_t bhBase = (size_t)b * HH + h;
        _Float16 hv = (_Float16)(acc[i][j][r] + bv);
        if (t == 0)      Qh[(bhBase * SS + s) * DK + d] = hv;
        else if (t == 1) Kh[(bhBase * SS + s) * DK + d] = hv;
        else             Vt[(bhBase * DK + d) * SS + s] = hv;
      }
    }
  }
}

// ---------------- Attention scores + softmax -------------------------------
// One block per (b,h, 16-query strip). Scores staged in 128KB dynamic LDS
// (CDNA5 WGP supports 320KB), row softmax via wave32 shuffle reductions,
// fp32 probabilities written to d_out.

__global__ void __launch_bounds__(256) attn_softmax(
    const _Float16* __restrict__ Qh, const _Float16* __restrict__ Kh,
    float* __restrict__ attn) {
  extern __shared__ float sm[];  // [16][SS]
  int bh = blockIdx.x / (SS / 16);
  int qt = blockIdx.x - bh * (SS / 16);
  const _Float16* Qp = Qh + (size_t)bh * SS * DK;
  const _Float16* Kp = Kh + (size_t)bh * SS * DK;
  float* out = attn + (size_t)bh * SS * SS + (size_t)qt * 16 * SS;
  int wave = threadIdx.x >> 5;
  int lane = threadIdx.x & 31;

  // Q tile fragments (d_k = 64 -> two K=32 steps), loaded once per wave.
  v16h a0 = load_a16(Qp, qt * 16, 0, DK);
  v16h a1 = load_a16(Qp, qt * 16, 32, DK);

  // 8 waves sweep 128 key tiles; B fragment = K rows (contiguous in [s,d] layout).
  for (int kt = wave; kt < SS / 16; kt += 8) {
    if (kt + 8 < SS / 16)
      pf(Kp + (size_t)((kt + 8) * 16 + (lane & 15)) * DK);
    v16h b0 = load_b16(Kp, kt * 16, 0, DK);
    v16h b1 = load_b16(Kp, kt * 16, 32, DK);
    v8f c = {};
    c = wmma16(a0, b0, c);
    c = wmma16(a1, b1, c);
    int col = kt * 16 + (lane & 15);
    int rbase = 8 * (lane >> 4);
#pragma unroll
    for (int r = 0; r < 8; ++r)
      sm[(rbase + r) * SS + col] = c[r] * 0.125f;  // 1/sqrt(64)
  }
  __syncthreads();

  for (int row = wave; row < 16; row += 8) {
    float* rp = sm + row * SS;
    float mx = -3.402823466e38f;
    for (int i = lane; i < SS; i += 32) mx = fmaxf(mx, rp[i]);
#pragma unroll
    for (int off = 16; off > 0; off >>= 1) mx = fmaxf(mx, __shfl_xor(mx, off, 32));
    float sum = 0.f;
    for (int i = lane; i < SS; i += 32) {
      float e = __expf(rp[i] - mx);
      rp[i] = e;
      sum += e;
    }
#pragma unroll
    for (int off = 16; off > 0; off >>= 1) sum += __shfl_xor(sum, off, 32);
    float inv = 1.f / sum;
    for (int i = lane; i < SS; i += 32) out[(size_t)row * SS + i] = rp[i] * inv;
  }
}

// ---------------- AV: attn[S,S] (fp32, from d_out) @ V -> Vals f16 [B,S,E] --
// Two 16-row M tiles x four 16-col N tiles per wave (dk = 64 = all N).

__global__ void __launch_bounds__(256) av_gemm(
    const float* __restrict__ attn, const _Float16* __restrict__ Vt,
    _Float16* __restrict__ Valh) {
  int wave = threadIdx.x >> 5;
  int gid = blockIdx.x * 8 + wave;
  int bh = gid / (SS / 32);            // 64 bh slices
  int mt = gid - bh * (SS / 32);       // 0..63, 32-row tiles
  int b = bh / HH, h = bh - b * HH;
  int row0 = mt * 32;
  const float* Ap = attn + (size_t)bh * SS * SS;
  const _Float16* Bp = Vt + (size_t)bh * DK * SS;  // [DK, S] N-major
  v8f c00 = {}, c01 = {}, c02 = {}, c03 = {};
  v8f c10 = {}, c11 = {}, c12 = {}, c13 = {};
  for (int k0 = 0; k0 < SS; k0 += 32) {
    v16h a0 = load_a32(Ap, row0,      k0, SS);
    v16h a1 = load_a32(Ap, row0 + 16, k0, SS);
    v16h b0 = load_b16(Bp,  0, k0, SS);
    v16h b1 = load_b16(Bp, 16, k0, SS);
    v16h b2 = load_b16(Bp, 32, k0, SS);
    v16h b3 = load_b16(Bp, 48, k0, SS);
    if (k0 + 128 < SS) {
      int lane = threadIdx.x & 31;
      pf(Ap + (size_t)(row0 + (lane & 15)) * SS + k0 + 128);
      pf(Bp + (size_t)(lane)*SS + k0 + 256);
    }
    c00 = wmma16(a0, b0, c00);
    c10 = wmma16(a1, b0, c10);
    c01 = wmma16(a0, b1, c01);
    c11 = wmma16(a1, b1, c11);
    c02 = wmma16(a0, b2, c02);
    c12 = wmma16(a1, b2, c12);
    c03 = wmma16(a0, b3, c03);
    c13 = wmma16(a1, b3, c13);
  }
  int lane = threadIdx.x & 31;
  int nl = lane & 15, hi = lane >> 4;
  v8f acc[2][4] = {{c00, c01, c02, c03}, {c10, c11, c12, c13}};
#pragma unroll
  for (int j = 0; j < 4; ++j) {
    int d = j * 16 + nl;
#pragma unroll
    for (int i = 0; i < 2; ++i) {
#pragma unroll
      for (int r = 0; r < 8; ++r) {
        int s = row0 + i * 16 + r + 8 * hi;
        Valh[((size_t)b * SS + s) * EE + h * DK + d] = (_Float16)acc[i][j][r];
      }
    }
  }
}

// ---------------- Output projection: Vals @ W_out + b_out -> fp32 ----------

__global__ void __launch_bounds__(256) out_gemm(
    const _Float16* __restrict__ Valh, const _Float16* __restrict__ WT,
    const float* __restrict__ bias, float* __restrict__ out) {
  int wave = threadIdx.x >> 5;
  int gid = blockIdx.x * 8 + wave;
  int mt = gid / (EE / 64);            // 0..255, 32-row tiles
  int ng = gid - mt * (EE / 64);       // 0..15
  int row0 = mt * 32;
  v8f c00 = {}, c01 = {}, c02 = {}, c03 = {};
  v8f c10 = {}, c11 = {}, c12 = {}, c13 = {};
  for (int k0 = 0; k0 < EE; k0 += 32) {
    v16h a0 = load_a16(Valh, row0,      k0, EE);
    v16h a1 = load_a16(Valh, row0 + 16, k0, EE);
    v16h b0 = load_b16(WT, ng * 64 +  0, k0, EE);
    v16h b1 = load_b16(WT, ng * 64 + 16, k0, EE);
    v16h b2 = load_b16(WT, ng * 64 + 32, k0, EE);
    v16h b3 = load_b16(WT, ng * 64 + 48, k0, EE);
    if (k0 + 256 < EE) {
      int lane = threadIdx.x & 31;
      pf(Valh + (size_t)(row0 + (lane & 15)) * EE + k0 + 256);
      pf(WT + (size_t)(ng * 64 + lane) * EE + k0 + 256);
    }
    c00 = wmma16(a0, b0, c00);
    c10 = wmma16(a1, b0, c10);
    c01 = wmma16(a0, b1, c01);
    c11 = wmma16(a1, b1, c11);
    c02 = wmma16(a0, b2, c02);
    c12 = wmma16(a1, b2, c12);
    c03 = wmma16(a0, b3, c03);
    c13 = wmma16(a1, b3, c13);
  }
  int lane = threadIdx.x & 31;
  int nl = lane & 15, hi = lane >> 4;
  v8f acc[2][4] = {{c00, c01, c02, c03}, {c10, c11, c12, c13}};
#pragma unroll
  for (int j = 0; j < 4; ++j) {
    int n = ng * 64 + j * 16 + nl;
    float bv = bias[n];
#pragma unroll
    for (int i = 0; i < 2; ++i) {
#pragma unroll
      for (int r = 0; r < 8; ++r) {
        int m = row0 + i * 16 + r + 8 * hi;
        out[(size_t)m * EE + n] = acc[i][j][r] + bv;
      }
    }
  }
}

// ---------------- Host launcher ---------------------------------------------

extern "C" void kernel_launch(void* const* d_in, const int* in_sizes, int n_in,
                              void* d_out, int out_size, void* d_ws, size_t ws_size,
                              hipStream_t stream) {
  const float* x     = (const float*)d_in[0];
  const float* W_qkv = (const float*)d_in[1];
  const float* b_qkv = (const float*)d_in[2];
  const float* W_out = (const float*)d_in[3];
  const float* b_out = (const float*)d_in[4];

  float* out  = (float*)d_out;                // [B,S,E]
  float* attn = out + (size_t)MM * EE;        // [B,H,S,S]

  char* ws = (char*)d_ws;
  _Float16* Xh    = (_Float16*)ws; ws += (size_t)MM * EE * 2;
  _Float16* WqkvT = (_Float16*)ws; ws += (size_t)N3 * EE * 2;
  _Float16* WoutT = (_Float16*)ws; ws += (size_t)EE * EE * 2;
  _Float16* Qh    = (_Float16*)ws; ws += (size_t)BB * HH * SS * DK * 2;
  _Float16* Kh    = (_Float16*)ws; ws += (size_t)BB * HH * SS * DK * 2;
  _Float16* Vt    = (_Float16*)ws; ws += (size_t)BB * HH * DK * SS * 2;
  _Float16* Valh  = (_Float16*)ws;

  // Allow 128KB dynamic LDS (CDNA5 WGP supports up to 320KB per workgroup).
  (void)hipFuncSetAttribute((const void*)attn_softmax,
                            hipFuncAttributeMaxDynamicSharedMemorySize,
                            16 * SS * (int)sizeof(float));

  cvt_f32_f16<<<4096, 256, 0, stream>>>(x, Xh, MM * EE);
  transpose_f32_f16<<<8192, 256, 0, stream>>>(W_qkv, WqkvT, EE, N3);
  transpose_f32_f16<<<4096, 256, 0, stream>>>(W_out, WoutT, EE, EE);

  // waves = (MM/32) * (N3/64) = 256*48 = 12288 -> 1536 blocks of 8 waves
  qkv_gemm<<<1536, 256, 0, stream>>>(Xh, WqkvT, b_qkv, Qh, Kh, Vt);

  attn_softmax<<<BB * HH * (SS / 16), 256, 16 * SS * sizeof(float), stream>>>(Qh, Kh, attn);

  // waves = (BB*HH) * (SS/32) = 64*64 = 4096 -> 512 blocks
  av_gemm<<<512, 256, 0, stream>>>(attn, Vt, Valh);

  // waves = (MM/32) * (EE/64) = 256*16 = 4096 -> 512 blocks
  out_gemm<<<512, 256, 0, stream>>>(Valh, WoutT, b_out, out);
}
